// Model_tv_51238959841614
// MI455X (gfx1250) — compile-verified, matrix operand order checked by reading, not run on hardware
//
#include <hip/hip_runtime.h>
#include <math.h>

typedef __attribute__((ext_vector_type(16))) __bf16 v16bf;
typedef __attribute__((ext_vector_type(2)))  __bf16 v2bf;
typedef __attribute__((ext_vector_type(8)))  float  v8f;

union BF16Frag {
    v16bf v;
    unsigned int u[8];
};

// ---- f32 pair -> packed bf16 ({hi[31:16], lo[31:16]}) --------------------
// v_perm_b32 selector 0x07060302: dst bytes = {a.b3, a.b2, b.b3, b.b2},
// where __builtin_amdgcn_perm(a, b, sel) indexes bytes of (a:b), b = low.

// Truncating pack: 1 VALU per pair.
__device__ __forceinline__ unsigned int packbf_trunc(float lo, float hi) {
#if __has_builtin(__builtin_amdgcn_cvt_pk_bf16_f32)
    v2bf p = __builtin_amdgcn_cvt_pk_bf16_f32(lo, hi);
    return __builtin_bit_cast(unsigned int, p);
#else
    return __builtin_amdgcn_perm(__float_as_uint(hi), __float_as_uint(lo),
                                 0x07060302u);
#endif
}

// Round-half-up pack (phase-2 weights, reused 100 steps): 3 VALU per pair.
__device__ __forceinline__ unsigned int packbf_rnd(float lo, float hi) {
#if __has_builtin(__builtin_amdgcn_cvt_pk_bf16_f32)
    v2bf p = __builtin_amdgcn_cvt_pk_bf16_f32(lo, hi);
    return __builtin_bit_cast(unsigned int, p);
#else
    const unsigned int xl = __float_as_uint(lo) + 0x8000u;
    const unsigned int xh = __float_as_uint(hi) + 0x8000u;
    return __builtin_amdgcn_perm(xh, xl, 0x07060302u);
#endif
}

__device__ __forceinline__ float fast_tanh(float x) {
#if __has_builtin(__builtin_amdgcn_tanhf)
    return __builtin_amdgcn_tanhf(x);
#else
    // tanh(x) = 1 - 2/(exp(2x)+1)  (v_exp_f32 + fast rcp path)
    const float e = __expf(2.0f * x);
    return 1.0f - __fdividef(2.0f, e + 1.0f);
#endif
}

// K index within a 32-wide chunk for element e of an A/B bf16 fragment:
// lanes 0-15 cover K = {0..7, 16..23}, lanes 16-31 cover K = {8..15, 24..31}
__device__ __forceinline__ int kOfElem(int e, int half) {
    return 16 * (e >> 3) + 8 * half + (e & 7);
}

// ---------------------------------------------------------------------------
// Phase 1: xp[t][h][b] = sum_d W_ih[s][h][d] * x[b][t][d] + b_ih[s][h] + b_hh[s][h]
// One wave per (t, b-tile); all 4 h-tiles per wave so each packed x-fragment
// (B operand) feeds 4 WMMAs. K-loop kept ROLLED: ~10 loads + 4 WMMAs in
// flight per iteration -> no spills, high occupancy for the HBM-bound phase.
// ---------------------------------------------------------------------------
__global__ void rnn_xproj(const float* __restrict__ x,
                          const float* __restrict__ W_ih,
                          const float* __restrict__ b_ih,
                          const float* __restrict__ b_hh,
                          float* __restrict__ xp) {
    const int lane = threadIdx.x & 31;
    const int wid  = blockIdx.x * (blockDim.x >> 5) + (threadIdx.x >> 5);
    const int bt = wid & 15;          // batch tile 0..15
    const int t  = wid >> 4;          // time step 0..999

    const int s    = t / 100;
    const int n    = lane & 15;
    const int half = lane >> 4;
    const int b    = bt * 16 + n;     // B-operand column / D column (batch)

    const float* xrow  = x + ((size_t)b * 1000 + t) * 256;
    const float* Wbase = W_ih + (size_t)s * 64 * 256;

    v8f acc[4] = {};
#pragma unroll 1
    for (int dk = 0; dk < 8; ++dk) {            // K chunks of 32 over D=256
        BF16Frag bb;
#pragma unroll
        for (int j = 0; j < 8; ++j) {
            const int k0 = 32 * dk + kOfElem(2 * j, half);     // k1 = k0 + 1
            bb.u[j] = packbf_trunc(xrow[k0], xrow[k0 + 1]);
        }
#pragma unroll
        for (int ht = 0; ht < 4; ++ht) {
            const float* Wrow = Wbase + (size_t)(ht * 16 + n) * 256;
            BF16Frag a;
#pragma unroll
            for (int j = 0; j < 8; ++j) {
                const int k0 = 32 * dk + kOfElem(2 * j, half);
                a.u[j] = packbf_trunc(Wrow[k0], Wrow[k0 + 1]);
            }
            acc[ht] = __builtin_amdgcn_wmma_f32_16x16x32_bf16(
                          false, a.v, false, bb.v, (short)0, acc[ht], false, false);
        }
    }

#pragma unroll
    for (int ht = 0; ht < 4; ++ht)
#pragma unroll
        for (int r = 0; r < 8; ++r) {
            const int h = ht * 16 + 8 * half + r;
            const float bias = b_ih[s * 64 + h] + b_hh[s * 64 + h];
            xp[((size_t)t * 64 + h) * 256 + b] = acc[ht][r] + bias;
        }
}

// ---------------------------------------------------------------------------
// Phase 2 helpers
// ---------------------------------------------------------------------------
__device__ __forceinline__ void load_c(const float* __restrict__ xp, int t,
                                       int half, int b, v8f (&c)[4]) {
#pragma unroll
    for (int mt = 0; mt < 4; ++mt)
#pragma unroll
        for (int r = 0; r < 8; ++r)
            c[mt][r] = xp[((size_t)t * 64 + mt * 16 + 8 * half + r) * 256 + b];
}

__device__ __forceinline__ void rnn_step(const BF16Frag (&A)[4][2],
                                         BF16Frag (&Bfrag)[2],
                                         const v8f (&c)[4],
                                         const float (&fcw)[4][8], float fcb,
                                         int t, int lane, int b,
                                         float* __restrict__ out) {
    v8f d[4];
    float partial = 0.f;
#pragma unroll
    for (int mt = 0; mt < 4; ++mt) {
        d[mt] = __builtin_amdgcn_wmma_f32_16x16x32_bf16(
                    false, A[mt][0].v, false, Bfrag[0].v, (short)0, c[mt], false, false);
        d[mt] = __builtin_amdgcn_wmma_f32_16x16x32_bf16(
                    false, A[mt][1].v, false, Bfrag[1].v, (short)0, d[mt], false, false);
        // tanh in place on the accumulator (no extra register array)
#pragma unroll
        for (int r = 0; r < 8; ++r) {
            const float hv = fast_tanh(d[mt][r]);
            d[mt][r] = hv;
            partial += fcw[mt][r] * hv;
        }
    }
    // Repack next-step B fragments: identical per-lane mapping, no shuffles.
    // Truncating pack: 1 v_perm_b32 per pair on the serial chain.
#pragma unroll
    for (int cc = 0; cc < 2; ++cc)
#pragma unroll
        for (int j = 0; j < 8; ++j) {
            const int e0 = 2 * j, e1 = 2 * j + 1;
            Bfrag[cc].u[j] = packbf_trunc(d[2 * cc + (e0 >> 3)][e0 & 7],
                                          d[2 * cc + (e1 >> 3)][e1 & 7]);
        }
    // Head: lane l and lane l^16 hold complementary halves of h for batch n.
    const float total = partial + __shfl_xor(partial, 16, 32);
    const float y = __fdividef(1.f, 1.f + __expf(-(total + fcb)));
    if (lane < 16) out[(size_t)b * 1000 + t] = y;
}

// ---------------------------------------------------------------------------
// Phase 2: serial recurrence. One wave per 16-batch tile (grid = 16 x 32 thr).
// D = W_hh x h^T : A = W_hh tiles (registers, per segment), B = h^T fragments.
// D layout == next-step B layout per lane -> pure register feedback.
// xp C-fragments are double-buffered (prefetch t+1 before step t's WMMAs).
// ---------------------------------------------------------------------------
__global__ void rnn_recur(const float* __restrict__ xp,
                          const float* __restrict__ W_hh,
                          const float* __restrict__ fc_w,
                          const float* __restrict__ fc_b,
                          float* __restrict__ out) {
    const int lane = threadIdx.x & 31;
    const int n    = lane & 15;
    const int half = lane >> 4;
    const int bt   = blockIdx.x;          // 0..15
    const int b    = bt * 16 + n;

    BF16Frag Bfrag[2];                    // h0 = 0
#pragma unroll
    for (int c = 0; c < 2; ++c)
#pragma unroll
        for (int j = 0; j < 8; ++j) Bfrag[c].u[j] = 0u;

    v8f cA[4], cB[4];
    load_c(xp, 0, half, b, cA);

    for (int s = 0; s < 10; ++s) {
        // Load W_hh[s] as 8 A-fragments (kept in registers for 100 steps)
        BF16Frag A[4][2];
        const float* W = W_hh + (size_t)s * 64 * 64;
#pragma unroll
        for (int mt = 0; mt < 4; ++mt)
#pragma unroll
            for (int c = 0; c < 2; ++c)
#pragma unroll
                for (int j = 0; j < 8; ++j) {
                    const int k0 = 32 * c + kOfElem(2 * j, half);
                    const float* row = W + (size_t)(mt * 16 + n) * 64;
                    A[mt][c].u[j] = packbf_rnd(row[k0], row[k0 + 1]);
                }

        float fcw[4][8];
#pragma unroll
        for (int mt = 0; mt < 4; ++mt)
#pragma unroll
            for (int r = 0; r < 8; ++r)
                fcw[mt][r] = fc_w[s * 64 + mt * 16 + 8 * half + r];
        const float fcb = fc_b[s];

        for (int tc = 0; tc < 100; tc += 2) {
            const int t  = s * 100 + tc;
            const int t1 = (t + 1 < 1000) ? t + 1 : 999;
            const int t2 = (t + 2 < 1000) ? t + 2 : 999;
            load_c(xp, t1, half, b, cB);                         // prefetch t+1
            rnn_step(A, Bfrag, cA, fcw, fcb, t, lane, b, out);   // consume t
            load_c(xp, t2, half, b, cA);                         // prefetch t+2
            rnn_step(A, Bfrag, cB, fcw, fcb, t + 1, lane, b, out);
        }
    }
}

extern "C" void kernel_launch(void* const* d_in, const int* in_sizes, int n_in,
                              void* d_out, int out_size, void* d_ws, size_t ws_size,
                              hipStream_t stream) {
    const float* x    = (const float*)d_in[0];
    const float* W_ih = (const float*)d_in[1];
    const float* W_hh = (const float*)d_in[2];
    const float* b_ih = (const float*)d_in[3];
    const float* b_hh = (const float*)d_in[4];
    const float* fc_w = (const float*)d_in[5];
    const float* fc_b = (const float*)d_in[6];
    float* out = (float*)d_out;
    float* xp  = (float*)d_ws;   // needs T*H*B*4 = 65.5 MB

    // Phase 1: 1000 * 16 = 16000 waves, 8 waves (256 thr) per block.
    rnn_xproj<<<2000, 256, 0, stream>>>(x, W_ih, b_ih, b_hh, xp);
    // Phase 2: one wave per 16-batch tile.
    rnn_recur<<<16, 32, 0, stream>>>(xp, W_hh, fc_w, fc_b, out);
}
